// SelfAttention_541165879640
// MI455X (gfx1250) — matrix operand order repaired and checked by reading, not compiled, not run on hardware
//
#include <hip/hip_runtime.h>
#include <hip/hip_bf16.h>
#include <stdint.h>

// ---------------------------------------------------------------------------
// Self-attention (B=16, C=256, H=W=64, N=4096, C8=32) for gfx1250 (MI455X).
// bf16 WMMA (v_wmma_f32_16x16x32_bf16), f32 accumulation, flash-style
// two-pass softmax so the N x N energy matrix is never materialized.
//
// Attention workgroup = 4 waves covering a 64-query strip; the 4 waves load
// identical K/V tiles each step (barrier-aligned) so V is pulled from L2 into
// the WGP cache once per workgroup instead of once per wave (4x L2 traffic
// reduction on the dominant 137-GFLOP P*V stage).
//
// Workspace layout (d_ws, ~40 MB):
//   qt : bf16 [B][N][32]   (transposed: B-operand friendly)
//   kt : bf16 [B][N][32]   (transposed: A-operand friendly)
//   vb : bf16 [B][256][N]  (row-major: A-operand friendly)
// ---------------------------------------------------------------------------

typedef __attribute__((ext_vector_type(16))) __bf16 v16bf;
typedef __attribute__((ext_vector_type(8)))  float  v8f;

union Frag {
  v16bf v;
  uint32_t u[8];
  uint4 q[2];
};

__device__ __forceinline__ uint32_t f32_to_bf16_rne(float f) {
  union { float f; uint32_t u; } c; c.f = f;
  uint32_t u = c.u;
  return ((u + 0x7FFFu + ((u >> 16) & 1u)) >> 16) & 0xFFFFu;
}
__device__ __forceinline__ uint32_t pack_bf16(float lo, float hi) {
  return f32_to_bf16_rne(lo) | (f32_to_bf16_rne(hi) << 16);
}

__device__ __forceinline__ v8f wmma_bf16(const Frag& A, const Frag& B, v8f C) {
  return __builtin_amdgcn_wmma_f32_16x16x32_bf16(false, A.v, false, B.v,
                                                 (short)0, C, false, false);
}

// ---------------------------------------------------------------------------
// Projection: Out[M x 4096] = W[M x 256] * X_b[256 x 4096] + bias, per batch.
// TRANSPOSED=1 -> out_t[n*M + o] (q/k, M=32); TRANSPOSED=0 -> out[o*4096 + n].
// One wave computes one 16x16 output tile; K-loop 256 in steps of 32.
// ---------------------------------------------------------------------------
template <int TRANSPOSED>
__global__ void proj_kernel(const float* __restrict__ W,
                            const float* __restrict__ bias,
                            const float* __restrict__ X,   // [B][256][4096] f32
                            uint16_t* __restrict__ out,    // bf16 bits
                            int M) {
  const int lane = threadIdx.x;
  const int half = lane >> 4;      // 0: lanes 0-15, 1: lanes 16-31
  const int l16  = lane & 15;
  const int n0 = blockIdx.x * 16;
  const int m0 = blockIdx.y * 16;
  const int b  = blockIdx.z;

  const float* Xb = X + (size_t)b * 256 * 4096;
  const int m = m0 + l16;
  const int n = n0 + l16;

  v8f acc = {};
  #pragma unroll
  for (int k0 = 0; k0 < 256; k0 += 32) {
    // A fragment (16x32 bf16): low half K=k0+0..15, high half K=k0+16..31.
    Frag A;
    {
      const int cbase = k0 + half * 16;
      const float4* p0 = (const float4*)(W + (size_t)m * 256 + cbase);
      float4 f0 = p0[0], f1 = p0[1];
      const float4* p1 = (const float4*)(W + (size_t)m * 256 + cbase + 8);
      float4 f2 = p1[0], f3 = p1[1];
      A.u[0] = pack_bf16(f0.x, f0.y); A.u[1] = pack_bf16(f0.z, f0.w);
      A.u[2] = pack_bf16(f1.x, f1.y); A.u[3] = pack_bf16(f1.z, f1.w);
      A.u[4] = pack_bf16(f2.x, f2.y); A.u[5] = pack_bf16(f2.z, f2.w);
      A.u[6] = pack_bf16(f3.x, f3.y); A.u[7] = pack_bf16(f3.z, f3.w);
    }
    // B fragment (32x16 bf16): lane column n, VGPR v packs rows (2v, 2v+1);
    // low half rows k0..k0+15, high half rows k0+16..k0+31.
    Frag Bf;
    {
      const int cbase = k0 + half * 16;
      #pragma unroll
      for (int v = 0; v < 8; ++v) {
        float f0 = Xb[(size_t)(cbase + 2 * v) * 4096 + n];
        float f1 = Xb[(size_t)(cbase + 2 * v + 1) * 4096 + n];
        Bf.u[v] = pack_bf16(f0, f1);
      }
    }
    acc = wmma_bf16(A, Bf, acc);
  }

  // D layout: VGPR r -> row m0 + half*8 + r, col n0 + l16.
  const int obase = m0 + half * 8;
  if (TRANSPOSED) {
    uint32_t pk0 = pack_bf16(acc[0] + bias[obase + 0], acc[1] + bias[obase + 1]);
    uint32_t pk1 = pack_bf16(acc[2] + bias[obase + 2], acc[3] + bias[obase + 3]);
    uint32_t pk2 = pack_bf16(acc[4] + bias[obase + 4], acc[5] + bias[obase + 5]);
    uint32_t pk3 = pack_bf16(acc[6] + bias[obase + 6], acc[7] + bias[obase + 7]);
    uint16_t* dst = out + ((size_t)b * 4096 + n) * M + obase;
    *(uint4*)dst = make_uint4(pk0, pk1, pk2, pk3);
  } else {
    uint16_t* dst = out + ((size_t)b * M + obase) * 4096 + n;
    #pragma unroll
    for (int r = 0; r < 8; ++r)
      dst[(size_t)r * 4096] = (uint16_t)f32_to_bf16_rne(acc[r] + bias[obase + r]);
  }
}

// ---------------------------------------------------------------------------
// Flash attention: 4 waves per workgroup; wave w handles query tile
// i0 = blockIdx.x*64 + w*16 of batch blockIdx.y. Waves share K/V addresses
// per step (WGP$ reuse). Pass 1: row max over all 4096 keys. Pass 2:
// recompute S, P = exp2, accumulate O[256 x 16] with 16 WMMAs / 32-key step,
// P transposed through a private 1 KB LDS slice.
// ---------------------------------------------------------------------------
__global__ void attn_kernel(const uint16_t* __restrict__ qt,  // [B][N][32]
                            const uint16_t* __restrict__ kt,  // [B][N][32]
                            const uint16_t* __restrict__ vb,  // [B][256][N]
                            const float* __restrict__ x,      // [B][256][N]
                            const float* __restrict__ gamma,  // [1]
                            float* __restrict__ outp) {       // [B][256][N]
  __shared__ uint32_t pk[4][16 * 16];  // per-wave slice: pk[i][t], t=j/2 pairs

  const int tid  = threadIdx.x;
  const int wave = tid >> 5;
  const int lane = tid & 31;
  const int half = lane >> 4;
  const int l16  = lane & 15;
  const int i0 = blockIdx.x * 64 + wave * 16;
  const int b  = blockIdx.y;
  const int N  = 4096;
  const float LOG2E = 1.4426950408889634f;

  const uint16_t* qtb = qt + (size_t)b * N * 32;
  const uint16_t* ktb = kt + (size_t)b * N * 32;
  const uint16_t* vbb = vb + (size_t)b * 256 * N;
  uint32_t* mypk = pk[wave];

  // Q as B operand (32x16): lane col i0+l16; low half c={0..7,16..23}, high +8.
  Frag Qf;
  {
    const uint16_t* src = qtb + (size_t)(i0 + l16) * 32 + half * 8;
    Qf.q[0] = *(const uint4*)(src);
    Qf.q[1] = *(const uint4*)(src + 16);
  }

  // ---- Pass 1: row max (per query i) ----
  float vmax = -3.0e38f;
  for (int jt = 0; jt < N; jt += 32) {
    Frag K0, K1;
    const uint16_t* k0p = ktb + (size_t)(jt + l16) * 32 + half * 8;
    K0.q[0] = *(const uint4*)(k0p);
    K0.q[1] = *(const uint4*)(k0p + 16);
    const uint16_t* k1p = ktb + (size_t)(jt + 16 + l16) * 32 + half * 8;
    K1.q[0] = *(const uint4*)(k1p);
    K1.q[1] = *(const uint4*)(k1p + 16);
    if (jt + 32 < N)
      __builtin_prefetch(k0p + 32 * 32, 0, 1);  // next K tile -> global_prefetch
    v8f z = {};
    v8f D0 = wmma_bf16(K0, Qf, z);   // S^T rows j = jt..jt+15
    v8f D1 = wmma_bf16(K1, Qf, z);   // S^T rows j = jt+16..jt+31
    #pragma unroll
    for (int r = 0; r < 8; ++r) {
      vmax = fmaxf(vmax, D0[r]);
      vmax = fmaxf(vmax, D1[r]);
    }
  }
  vmax = fmaxf(vmax, __shfl_xor(vmax, 16, 32));  // combine lane halves (same i)
  const float m2 = vmax * LOG2E;

  // ---- Pass 2: P = exp, O += V * P ----
  float lsum = 0.0f;
  v8f O[16];
  #pragma unroll
  for (int ct = 0; ct < 16; ++ct) O[ct] = (v8f){};

  for (int jt = 0; jt < N; jt += 32) {
    Frag K0, K1;
    const uint16_t* k0p = ktb + (size_t)(jt + l16) * 32 + half * 8;
    K0.q[0] = *(const uint4*)(k0p);
    K0.q[1] = *(const uint4*)(k0p + 16);
    const uint16_t* k1p = ktb + (size_t)(jt + 16 + l16) * 32 + half * 8;
    K1.q[0] = *(const uint4*)(k1p);
    K1.q[1] = *(const uint4*)(k1p + 16);
    v8f z = {};
    v8f D0 = wmma_bf16(K0, Qf, z);
    v8f D1 = wmma_bf16(K1, Qf, z);

    float p0[8], p1[8];
    #pragma unroll
    for (int r = 0; r < 8; ++r) {
      p0[r] = exp2f(D0[r] * LOG2E - m2);
      p1[r] = exp2f(D1[r] * LOG2E - m2);
      lsum += p0[r] + p1[r];
    }

    // Transpose P via LDS into B-operand layout. D0 low-half lane holds
    // (j=r, i=l16) -> t=0..3; high half (j=8+r) -> t=4..7; D1 -> t=8..15.
    {
      uint4 s0 = make_uint4(pack_bf16(p0[0], p0[1]), pack_bf16(p0[2], p0[3]),
                            pack_bf16(p0[4], p0[5]), pack_bf16(p0[6], p0[7]));
      uint4 s1 = make_uint4(pack_bf16(p1[0], p1[1]), pack_bf16(p1[2], p1[3]),
                            pack_bf16(p1[4], p1[5]), pack_bf16(p1[6], p1[7]));
      *(uint4*)&mypk[l16 * 16 + half * 4]     = s0;  // t = 0..3 / 4..7
      *(uint4*)&mypk[l16 * 16 + 8 + half * 4] = s1;  // t = 8..11 / 12..15
    }
    // Orders this wave's LDS store->load (dscnt) and keeps the 4 waves
    // stepping together so their identical V loads hit the WGP cache.
    __syncthreads();

    Frag Pf;  // B operand: low lanes t=0..3,8..11 (j=0..7,16..23), high t=4..7,12..15
    {
      const uint32_t* src = &mypk[l16 * 16 + half * 4];
      Pf.q[0] = *(const uint4*)(src);
      Pf.q[1] = *(const uint4*)(src + 8);
    }

    // O[ct] += V_tile(16c x 32j) * P(32j x 16i); all 4 waves read the same V.
    #pragma unroll
    for (int ct = 0; ct < 16; ++ct) {
      Frag Vf;
      const uint16_t* vp = vbb + (size_t)(ct * 16 + l16) * N + jt + half * 8;
      Vf.q[0] = *(const uint4*)(vp);
      Vf.q[1] = *(const uint4*)(vp + 16);
      if (jt + 32 < N)
        __builtin_prefetch(vp + 32, 0, 1);  // next j-tile of this V row
      O[ct] = wmma_bf16(Vf, Pf, O[ct]);
    }
  }

  // ---- Epilogue: out = gamma * O / lsum + x ----
  lsum += __shfl_xor(lsum, 16, 32);
  const float inv = 1.0f / lsum;
  const float g = gamma[0];
  const float* xb = x + (size_t)b * 256 * N;
  float* ob = outp + (size_t)b * 256 * N;
  const int col = i0 + l16;
  #pragma unroll
  for (int ct = 0; ct < 16; ++ct) {
    const int crow = ct * 16 + half * 8;
    #pragma unroll
    for (int r = 0; r < 8; ++r) {
      size_t idx = (size_t)(crow + r) * N + col;
      ob[idx] = g * (O[ct][r] * inv) + xb[idx];
    }
  }
}

// ---------------------------------------------------------------------------
extern "C" void kernel_launch(void* const* d_in, const int* in_sizes, int n_in,
                              void* d_out, int out_size, void* d_ws, size_t ws_size,
                              hipStream_t stream) {
  (void)in_sizes; (void)n_in; (void)out_size; (void)ws_size;
  const float* x     = (const float*)d_in[0];
  const float* Wq    = (const float*)d_in[1];
  const float* bq    = (const float*)d_in[2];
  const float* Wk    = (const float*)d_in[3];
  const float* bk    = (const float*)d_in[4];
  const float* Wv    = (const float*)d_in[5];
  const float* bv    = (const float*)d_in[6];
  const float* gamma = (const float*)d_in[7];
  float* out = (float*)d_out;

  const size_t B = 16, N = 4096;
  uint16_t* qt = (uint16_t*)d_ws;                 //  4 MB
  uint16_t* kt = qt + B * N * 32;                 //  4 MB
  uint16_t* vb = kt + B * N * 32;                 // 32 MB

  hipLaunchKernelGGL((proj_kernel<1>), dim3(256, 2, 16),  dim3(32), 0, stream, Wq, bq, x, qt, 32);
  hipLaunchKernelGGL((proj_kernel<1>), dim3(256, 2, 16),  dim3(32), 0, stream, Wk, bk, x, kt, 32);
  hipLaunchKernelGGL((proj_kernel<0>), dim3(256, 16, 16), dim3(32), 0, stream, Wv, bv, x, vb, 256);
  hipLaunchKernelGGL(attn_kernel,      dim3(64, 16),      dim3(128), 0, stream, qt, kt, vb, x, gamma, out);
}
